// STGNN_1709396984379
// MI455X (gfx1250) — matrix operand order repaired and checked by reading, not compiled, not run on hardware
//
#include <hip/hip_runtime.h>

#define B_  64
#define T_  512
#define N_  14
#define F_  16
#define H_  128
#define G4_ 512          // 4*H
#define M_  (B_*N_)      // 896

typedef __attribute__((ext_vector_type(16))) _Float16 v16h;
typedef __attribute__((ext_vector_type(8)))  _Float16 v8h;
typedef __attribute__((ext_vector_type(8)))  float    v8f;

// Fast transcendentals: single v_exp_f32 / v_rcp_f32 / v_tanh_f32 trans ops
// (co-execute with XDL WMMA per ISA; avoids IEEE v_div_scale sequences).
__device__ __forceinline__ float sigm_f(float x) {
    return __builtin_amdgcn_rcpf(1.0f + __expf(-x));
}
__device__ __forceinline__ float tanh_f(float x) {
#if __has_builtin(__builtin_amdgcn_tanhf)
    return __builtin_amdgcn_tanhf(x);
#else
    float e = __expf(2.0f * x);
    return 1.0f - 2.0f * __builtin_amdgcn_rcpf(e + 1.0f);
#endif
}

__device__ __forceinline__ v16h cat8(v8h a, v8h b) {
    return __builtin_shufflevector(a, b, 0,1,2,3,4,5,6,7,8,9,10,11,12,13,14,15);
}

// ---------------------------------------------------------------------------
// Prep: f32 weights -> f16 (W_hh [512x128], W_ih padded to [512x32]),
// bias = b_ih + b_hh. One pass, 65536 threads.
// ---------------------------------------------------------------------------
__global__ void stgnn_prep_kernel(const float* __restrict__ Whh,
                                  const float* __restrict__ Wih,
                                  const float* __restrict__ bih,
                                  const float* __restrict__ bhh,
                                  _Float16* __restrict__ WhhH,
                                  _Float16* __restrict__ WihH,
                                  float* __restrict__ bias) {
    int i = blockIdx.x * blockDim.x + threadIdx.x;      // 0 .. 65535
    if (i < G4_ * H_) WhhH[i] = (_Float16)Whh[i];
    if (i < G4_ * 32) {
        int row = i >> 5, col = i & 31;
        WihH[i] = (col < F_) ? (_Float16)Wih[row * F_ + col] : (_Float16)0.0f;
    }
    if (i < G4_) bias[i] = bih[i] + bhh[i];
}

// ---------------------------------------------------------------------------
// LSTM: one workgroup = 16 sequences, 8 waves, full T loop.
// Wave w owns hidden columns [w*16, w*16+16) of all four gates.
// 20 chained v_wmma_f32_16x16x32_f16 per wave per step; bias folded into the
// accumulator init; ping-pong h buffer -> single barrier per step.
// ---------------------------------------------------------------------------
__global__ __launch_bounds__(256) void stgnn_lstm_kernel(
        const float*    __restrict__ x,      // [B,T,N,F]
        const _Float16* __restrict__ WhhH,   // [512,128] f16
        const _Float16* __restrict__ WihH,   // [512,32]  f16 (K-padded)
        const float*    __restrict__ bias,   // [512]
        float*          __restrict__ hlast)  // [M,128]
{
    __shared__ _Float16 hbuf0[16 * H_];      // ping
    __shared__ _Float16 hbuf1[16 * H_];      // pong

    const int tid  = threadIdx.x;
    const int wave = tid >> 5;
    const int lane = tid & 31;
    const int r    = lane & 15;              // A row / B col / C col within tile
    const int hi   = (lane >> 4) & 1;        // lane half
    const int mtile = blockIdx.x;

    // ---- persistent B fragments (weights), loaded once -------------------
    // B 32x16 f16 layout: lane n=lane&15 holds K hi*8..hi*8+7 then 16+hi*8..16+hi*8+7
    v16h Bh[4][4];
    v16h Bx[4];
#pragma unroll
    for (int g = 0; g < 4; ++g) {
        const int row = g * H_ + wave * 16 + r;     // W row = gate column
        const _Float16* wp = WihH + row * 32 + hi * 8;
        Bx[g] = cat8(*(const v8h*)wp, *(const v8h*)(wp + 16));
#pragma unroll
        for (int kc = 0; kc < 4; ++kc) {
            const _Float16* p = WhhH + row * H_ + kc * 32 + hi * 8;
            Bh[g][kc] = cat8(*(const v8h*)p, *(const v8h*)(p + 16));
        }
    }
    float biasv[4];
#pragma unroll
    for (int g = 0; g < 4; ++g) biasv[g] = bias[g * H_ + wave * 16 + r];

    // ---- per-lane x pointer (sequence m = b*N + n) -----------------------
    const int m  = mtile * 16 + r;
    const int bb = m / N_;
    const int nn = m % N_;
    const float* xp = x + ((size_t)bb * T_ * N_ + nn) * F_ + hi * 8;

    // ---- init state ------------------------------------------------------
    for (int idx = tid; idx < 16 * H_; idx += 256) hbuf0[idx] = (_Float16)0.0f;
    v8f cst = {0.f, 0.f, 0.f, 0.f, 0.f, 0.f, 0.f, 0.f};
    float hout[8];
#pragma unroll
    for (int e = 0; e < 8; ++e) hout[e] = 0.0f;
    __syncthreads();

    _Float16* hread  = hbuf0;
    _Float16* hwrite = hbuf1;

    for (int t = 0; t < T_; ++t) {
        // A_x fragment: K = feature (0..15 valid, 16..31 zero-pad)
        float4 x0 = *(const float4*)xp;
        float4 x1 = *(const float4*)(xp + 4);
        xp += N_ * F_;
        __builtin_prefetch(xp, 0, 1);             // global_prefetch_b8 next step
        v16h Ax;
        Ax[0] = (_Float16)x0.x; Ax[1] = (_Float16)x0.y;
        Ax[2] = (_Float16)x0.z; Ax[3] = (_Float16)x0.w;
        Ax[4] = (_Float16)x1.x; Ax[5] = (_Float16)x1.y;
        Ax[6] = (_Float16)x1.z; Ax[7] = (_Float16)x1.w;
#pragma unroll
        for (int e = 8; e < 16; ++e) Ax[e] = (_Float16)0.0f;

        // A_h fragments from LDS (16-bit A layout: elems0-7 = K hi*8.., 8-15 = K 16+hi*8..)
        v16h Ah[4];
#pragma unroll
        for (int kc = 0; kc < 4; ++kc) {
            const _Float16* p = hread + r * H_ + kc * 32 + hi * 8;
            Ah[kc] = cat8(*(const v8h*)p, *(const v8h*)(p + 16));
        }

        // gates = bias + x@W_ih^T + h@W_hh^T   (bias pre-loaded into C)
        v8f acc[4];
#pragma unroll
        for (int g = 0; g < 4; ++g) {
            const float bv = biasv[g];
            v8f a = {bv, bv, bv, bv, bv, bv, bv, bv};
            a = __builtin_amdgcn_wmma_f32_16x16x32_f16(false, Ax, false, Bx[g],
                                                       (short)0, a, false, false);
#pragma unroll
            for (int kc = 0; kc < 4; ++kc)
                a = __builtin_amdgcn_wmma_f32_16x16x32_f16(false, Ah[kc], false, Bh[g][kc],
                                                           (short)0, a, false, false);
            acc[g] = a;
        }

        // LSTM cell update (torch gate order i,f,g,o) — all in-wave registers
#pragma unroll
        for (int e = 0; e < 8; ++e) {
            float iv = sigm_f(acc[0][e]);
            float fv = sigm_f(acc[1][e]);
            float gv = tanh_f(acc[2][e]);
            float ov = sigm_f(acc[3][e]);
            float cv = fv * cst[e] + iv * gv;
            cst[e] = cv;
            hout[e] = ov * tanh_f(cv);
        }

        // ping-pong: write new h to the other buffer; reads of hread were
        // consumed by this wave's WMMAs already, so one barrier suffices.
#pragma unroll
        for (int e = 0; e < 8; ++e)          // C layout: vgpr e -> row e + hi*8, col r
            hwrite[(e + hi * 8) * H_ + wave * 16 + r] = (_Float16)hout[e];

        _Float16* tmp = hread; hread = hwrite; hwrite = tmp;
        __syncthreads();                     // new h visible for next step
    }

    // h_last (f32, full precision from registers)
    float* hl = hlast + (size_t)mtile * 16 * H_;
#pragma unroll
    for (int e = 0; e < 8; ++e)
        hl[(e + hi * 8) * H_ + wave * 16 + r] = hout[e];
}

// ---------------------------------------------------------------------------
// GCN + heads: one workgroup per batch element (tiny: 14 nodes).
// ---------------------------------------------------------------------------
__global__ __launch_bounds__(64) void stgnn_gcn_kernel(
        const float* __restrict__ hlast,   // [M,128]
        const float* __restrict__ adj,     // [14,14]
        const float* __restrict__ W_gcn,   // [64,128]
        const float* __restrict__ b_gcn,   // [64]
        const float* __restrict__ W_out,   // [1,64]
        const float* __restrict__ b_out,   // [1]
        const float* __restrict__ W_lin1,  // [28,14]
        const float* __restrict__ b_lin1,  // [28]
        float* __restrict__ out)           // [B,28]
{
    __shared__ float dinv[N_];
    __shared__ float An[N_ * N_];
    __shared__ float sh[N_ * H_];
    __shared__ float xg[N_ * H_];
    __shared__ float t2[N_ * 64];
    __shared__ float s[N_];

    const int b = blockIdx.x;
    const int tid = threadIdx.x;

    if (tid < N_) {
        float sum = 1.0f;                              // self loop
        for (int j = 0; j < N_; ++j) sum += adj[tid * N_ + j];
        dinv[tid] = rsqrtf(sum);
    }
    __syncthreads();
    for (int idx = tid; idx < N_ * N_; idx += 64) {
        int i = idx / N_, j = idx % N_;
        float a = adj[i * N_ + j] + (i == j ? 1.0f : 0.0f);
        An[idx] = dinv[i] * a * dinv[j];
    }
    for (int idx = tid; idx < N_ * H_; idx += 64)
        sh[idx] = hlast[(size_t)(b * N_) * H_ + idx];
    __syncthreads();

    for (int idx = tid; idx < N_ * H_; idx += 64) {    // A_norm @ lstm_out
        int i = idx / H_, k = idx % H_;
        float acc = 0.0f;
        for (int j = 0; j < N_; ++j) acc += An[i * N_ + j] * sh[j * H_ + k];
        xg[idx] = acc;
    }
    __syncthreads();

    for (int idx = tid; idx < N_ * 64; idx += 64) {    // relu(xg @ W_gcn^T + b)
        int i = idx / 64, o = idx % 64;
        float acc = b_gcn[o];
        for (int k = 0; k < H_; ++k) acc += xg[i * H_ + k] * W_gcn[o * H_ + k];
        t2[idx] = fmaxf(acc, 0.0f);
    }
    __syncthreads();

    if (tid < N_) {                                    // @ W_out^T + b_out
        float acc = b_out[0];
        for (int o = 0; o < 64; ++o) acc += t2[tid * 64 + o] * W_out[o];
        s[tid] = acc;
    }
    __syncthreads();

    if (tid < 28) {                                    // sigmoid(s @ W_lin1^T + b)
        float acc = b_lin1[tid];
        for (int n = 0; n < N_; ++n) acc += s[n] * W_lin1[tid * N_ + n];
        out[b * 28 + tid] = sigm_f(acc);
    }
}

// ---------------------------------------------------------------------------
extern "C" void kernel_launch(void* const* d_in, const int* in_sizes, int n_in,
                              void* d_out, int out_size, void* d_ws, size_t ws_size,
                              hipStream_t stream) {
    const float* x      = (const float*)d_in[0];
    const float* adj    = (const float*)d_in[1];
    const float* W_ih   = (const float*)d_in[2];
    const float* W_hh   = (const float*)d_in[3];
    const float* b_ih   = (const float*)d_in[4];
    const float* b_hh   = (const float*)d_in[5];
    const float* W_gcn  = (const float*)d_in[6];
    const float* b_gcn  = (const float*)d_in[7];
    const float* W_out  = (const float*)d_in[8];
    const float* b_out  = (const float*)d_in[9];
    const float* W_lin1 = (const float*)d_in[10];
    const float* b_lin1 = (const float*)d_in[11];
    float* out = (float*)d_out;

    char* ws = (char*)d_ws;
    _Float16* WhhH  = (_Float16*)(ws);                               // 131072 B
    _Float16* WihH  = (_Float16*)(ws + 131072);                      //  32768 B
    float*    bias  = (float*)   (ws + 131072 + 32768);              //   2048 B
    float*    hlast = (float*)   (ws + 131072 + 32768 + 2048);       // 458752 B

    stgnn_prep_kernel<<<256, 256, 0, stream>>>(W_hh, W_ih, b_ih, b_hh,
                                               WhhH, WihH, bias);
    stgnn_lstm_kernel<<<M_ / 16, 256, 0, stream>>>(x, WhhH, WihH, bias, hlast);
    stgnn_gcn_kernel<<<B_, 64, 0, stream>>>(hlast, adj, W_gcn, b_gcn,
                                            W_out, b_out, W_lin1, b_lin1, out);
}
